// GNNComponent_75677323755588
// MI455X (gfx1250) — compile-verified, matrix operand order checked by reading, not run on hardware
//
#include <hip/hip_runtime.h>

#define N_NODES 100000
#define F_IN    128
#define HD      64
#define NE      3200000
#define NET     (NE + N_NODES)   // edges + self loops
#define NEG_SLOPE 0.2f
#define LN_EPS  1e-5f

typedef __attribute__((ext_vector_type(16))) __bf16 v16bf;
typedef __attribute__((ext_vector_type(8)))  float  v8f;
typedef unsigned short ushort_t;

// ---------- helpers ----------
__device__ __forceinline__ ushort_t f2bf(float f) {
    unsigned u = __float_as_uint(f);
    unsigned r = (u + 0x7fffu + ((u >> 16) & 1u)) >> 16;
    return (ushort_t)r;
}
__device__ __forceinline__ int enc_f(float f) {
    int i = __float_as_int(f);
    return i >= 0 ? i : (i ^ 0x7fffffff);
}
__device__ __forceinline__ float dec_f(int i) {
    return __int_as_float(i >= 0 ? i : (i ^ 0x7fffffff));
}
__device__ __forceinline__ float lrelu(float v) { return v > 0.f ? v : NEG_SLOPE * v; }

// ---------- generic fill ----------
__global__ void fill_k(unsigned* p, unsigned v, long long n) {
    long long i = (long long)blockIdx.x * blockDim.x + threadIdx.x;
    if (i < n) p[i] = v;
}

// ---------- f32 -> bf16 convert ----------
__global__ void cvt_bf_k(const float* __restrict__ s, ushort_t* __restrict__ d, long long n) {
    long long i = (long long)blockIdx.x * blockDim.x + threadIdx.x;
    if (i < n) d[i] = f2bf(s[i]);
}

// ---------- pack f32 weight [K,64] into bf16 WMMA B-fragments ----------
// frag layout: [(tn*KC + kc)*32 + lane] * 16 bf16, lane-local 16 elems contiguous.
// Element j of lane's fragment covers K index kc*32 + j + (j&8) + (lane>=16 ? 8:0),
// column tn*16 + (lane&15)  -- matches the 16-bit B operand VGPR layout.
__global__ void pack_bfrag_k(const float* __restrict__ W, ushort_t* __restrict__ frag, int K) {
    int t = blockIdx.x * blockDim.x + threadIdx.x;
    int KC = K >> 5;
    int total = 4 * KC * 32;
    if (t >= total) return;
    int lane = t & 31;
    int kc   = (t >> 5) % KC;
    int tn   = (t >> 5) / KC;
    int half8 = (lane >> 4) << 3;
    int col = tn * 16 + (lane & 15);
    ushort_t* o = frag + (size_t)t * 16;
#pragma unroll
    for (int j = 0; j < 16; ++j) {
        int kk = kc * 32 + j + (j & 8) + half8;
        o[j] = f2bf(W[(size_t)kk * HD + col]);
    }
}

// ---------- bf16 WMMA GEMM, one wave per 16x64 row panel ----------
// C[M,64] = A[M,K](bf16 row-major) * B[K,64] (pre-packed fragments), f32 accum.
template <int K>
__global__ void gemm_rowpanel_wmma(const ushort_t* __restrict__ A,
                                   const ushort_t* __restrict__ Bfrag,
                                   float* __restrict__ C, int M) {
    constexpr int KC = K / 32;
    int wave = (blockIdx.x * blockDim.x + threadIdx.x) >> 5;
    int lane = threadIdx.x & 31;
    if (wave >= (M >> 4)) return;
    int half = lane >> 4;
    int mrow = lane & 15;

    v8f acc[4] = {};
    const ushort_t* ap0 = A + (size_t)(wave * 16 + mrow) * K + half * 8;
#pragma unroll
    for (int kc = 0; kc < KC; ++kc) {
        union { uint4 q[2]; v16bf v; } a;
        a.q[0] = *(const uint4*)(ap0 + kc * 32);
        a.q[1] = *(const uint4*)(ap0 + kc * 32 + 16);
#pragma unroll
        for (int tn = 0; tn < 4; ++tn) {
            union { uint4 q[2]; v16bf v; } b;
            const uint4* bp = (const uint4*)(Bfrag + ((size_t)(tn * KC + kc) * 32 + lane) * 16);
            b.q[0] = bp[0];
            b.q[1] = bp[1];
            acc[tn] = __builtin_amdgcn_wmma_f32_16x16x32_bf16(false, a.v, false, b.v,
                                                              (short)0, acc[tn], false, false);
        }
    }
    float* crow = C + (size_t)(wave * 16 + half * 8) * HD;
#pragma unroll
    for (int tn = 0; tn < 4; ++tn)
#pragma unroll
        for (int r = 0; r < 8; ++r)
            crow[(size_t)r * HD + tn * 16 + mrow] = acc[tn][r];
}

// ---------- per-node attention dot products ----------
__global__ void att_dot_k(const float* __restrict__ h, const float* __restrict__ av,
                          const float* __restrict__ dv, float* __restrict__ as_o,
                          float* __restrict__ ad_o, int n_nodes) {
    int n = blockIdx.x * blockDim.x + threadIdx.x;
    if (n >= n_nodes) return;
#pragma unroll
    for (int hh = 0; hh < 2; ++hh) {
        float s0 = 0.f, s1 = 0.f;
#pragma unroll 8
        for (int c = 0; c < 32; ++c) {
            float v = h[(size_t)n * HD + hh * 32 + c];
            s0 += v * av[hh * 32 + c];
            s1 += v * dv[hh * 32 + c];
        }
        as_o[n * 2 + hh] = s0;
        ad_o[n * 2 + hh] = s1;
    }
}

// ---------- edge pass 1: segment max ----------
__global__ void edge_max_k(const int* __restrict__ ei, const float* __restrict__ as_,
                           const float* __restrict__ ad_, int* __restrict__ amax) {
    long long e = (long long)blockIdx.x * blockDim.x + threadIdx.x;
    if (e >= NET) return;
    int s, d;
    if (e < NE) { s = ei[e]; d = ei[(long long)NE + e]; } else { s = d = (int)(e - NE); }
#pragma unroll
    for (int h = 0; h < 2; ++h) {
        float a = lrelu(as_[s * 2 + h] + ad_[d * 2 + h]);
        atomicMax(&amax[d * 2 + h], enc_f(a));
    }
}

// ---------- edge pass 2: exp + segment sum ----------
__global__ void edge_exp_k(const int* __restrict__ ei, const float* __restrict__ as_,
                           const float* __restrict__ ad_, const int* __restrict__ amax,
                           float* __restrict__ ea, float* __restrict__ den) {
    long long e = (long long)blockIdx.x * blockDim.x + threadIdx.x;
    if (e >= NET) return;
    int s, d;
    if (e < NE) { s = ei[e]; d = ei[(long long)NE + e]; } else { s = d = (int)(e - NE); }
#pragma unroll
    for (int h = 0; h < 2; ++h) {
        float a = lrelu(as_[s * 2 + h] + ad_[d * 2 + h]);
        float ex = __expf(a - dec_f(amax[d * 2 + h]));
        ea[(size_t)e * 2 + h] = ex;
        atomicAdd(&den[d * 2 + h], ex);
    }
}

// ---------- edge pass 3: message scatter-add ----------
__global__ void edge_msg_k(const int* __restrict__ ei, const float* __restrict__ ea,
                           const float* __restrict__ den, const float* __restrict__ h,
                           float* __restrict__ agg) {
    long long idx = (long long)blockIdx.x * blockDim.x + threadIdx.x;
    if (idx >= (long long)NET * HD) return;
    long long e = idx >> 6;
    int c = (int)(idx & 63);
    int hh = c >> 5;
    int s, d;
    if (e < NE) { s = ei[e]; d = ei[(long long)NE + e]; } else { s = d = (int)(e - NE); }
    float coef = ea[(size_t)e * 2 + hh] / den[d * 2 + hh];
    atomicAdd(&agg[(size_t)d * HD + c], h[(size_t)s * HD + c] * coef);
}

// ---------- bias + layernorm (+residual) + ELU, optional bf16 copy; one wave/node ----------
__global__ void ln_elu_k(const float* __restrict__ agg, const float* __restrict__ bias,
                         const float* __restrict__ g, const float* __restrict__ be,
                         const float* __restrict__ res, float* __restrict__ xout,
                         ushort_t* __restrict__ bfout, int n_nodes) {
    int n = (blockIdx.x * blockDim.x + threadIdx.x) >> 5;
    int lane = threadIdx.x & 31;
    if (n >= n_nodes) return;
    float v0 = agg[(size_t)n * HD + lane] + bias[lane];
    float v1 = agg[(size_t)n * HD + 32 + lane] + bias[32 + lane];
    float s = v0 + v1;
    for (int m = 16; m >= 1; m >>= 1) s += __shfl_xor(s, m, 32);
    float mu = s * (1.f / 64.f);
    float d0 = v0 - mu, d1 = v1 - mu;
    float q = d0 * d0 + d1 * d1;
    for (int m = 16; m >= 1; m >>= 1) q += __shfl_xor(q, m, 32);
    float inv = rsqrtf(q * (1.f / 64.f) + LN_EPS);
    float y0 = d0 * inv * g[lane] + be[lane];
    float y1 = d1 * inv * g[32 + lane] + be[32 + lane];
    if (res) {
        y0 += res[(size_t)n * HD + lane];
        y1 += res[(size_t)n * HD + 32 + lane];
    }
    y0 = y0 > 0.f ? y0 : (__expf(y0) - 1.f);
    y1 = y1 > 0.f ? y1 : (__expf(y1) - 1.f);
    xout[(size_t)n * HD + lane] = y0;
    xout[(size_t)n * HD + 32 + lane] = y1;
    if (bfout) {
        bfout[(size_t)n * HD + lane] = f2bf(y0);
        bfout[(size_t)n * HD + 32 + lane] = f2bf(y1);
    }
}

// ---------- node score: tanh(x2@Wa1+ba1)@Wa2 + ba2 ; one wave/node ----------
__global__ void score_k(const float* __restrict__ x2, const float* __restrict__ Wa1,
                        const float* __restrict__ ba1, const float* __restrict__ Wa2,
                        const float* __restrict__ ba2, float* __restrict__ score, int n_nodes) {
    int n = (blockIdx.x * blockDim.x + threadIdx.x) >> 5;
    int lane = threadIdx.x & 31;
    if (n >= n_nodes) return;
    float acc = ba1[lane];
#pragma unroll 8
    for (int k = 0; k < HD; ++k) acc += x2[(size_t)n * HD + k] * Wa1[k * 32 + lane];
    float t = tanhf(acc) * Wa2[lane];
    for (int m = 16; m >= 1; m >>= 1) t += __shfl_xor(t, m, 32);
    if (lane == 0) score[n] = t + ba2[0];
}

// ---------- global softmax reductions ----------
__global__ void smax_k(const float* __restrict__ score, int* __restrict__ smax, int n_nodes) {
    float v = -3.4e38f;
    for (int i = blockIdx.x * blockDim.x + threadIdx.x; i < n_nodes; i += gridDim.x * blockDim.x)
        v = fmaxf(v, score[i]);
    for (int m = 16; m >= 1; m >>= 1) v = fmaxf(v, __shfl_xor(v, m, 32));
    if ((threadIdx.x & 31) == 0) atomicMax(smax, enc_f(v));
}
__global__ void ssum_k(const float* __restrict__ score, const int* __restrict__ smax,
                       float* __restrict__ ssum, int n_nodes) {
    float mx = dec_f(*smax);
    float v = 0.f;
    for (int i = blockIdx.x * blockDim.x + threadIdx.x; i < n_nodes; i += gridDim.x * blockDim.x)
        v += __expf(score[i] - mx);
    for (int m = 16; m >= 1; m >>= 1) v += __shfl_xor(v, m, 32);
    if ((threadIdx.x & 31) == 0) atomicAdd(ssum, v);
}

// ---------- pooled = sum_n x2[n]*softmax(score)[n]; blockDim = 64 ----------
__global__ void pool_k(const float* __restrict__ x2, const float* __restrict__ score,
                       const int* __restrict__ smax, const float* __restrict__ ssum,
                       float* __restrict__ out, int n_nodes) {
    int j = threadIdx.x;
    float mx = dec_f(*smax);
    float inv = 1.f / *ssum;
    float acc = 0.f;
    for (int n = blockIdx.x; n < n_nodes; n += gridDim.x)
        acc += x2[(size_t)n * HD + j] * __expf(score[n] - mx);
    atomicAdd(&out[j], acc * inv);
}

// ==================================================================
extern "C" void kernel_launch(void* const* d_in, const int* in_sizes, int n_in,
                              void* d_out, int out_size, void* d_ws, size_t ws_size,
                              hipStream_t stream) {
    const float* x    = (const float*)d_in[0];
    const int*   ei   = (const int*)d_in[1];
    const float* W1   = (const float*)d_in[2];
    const float* as1  = (const float*)d_in[3];
    const float* ad1  = (const float*)d_in[4];
    const float* b1   = (const float*)d_in[5];
    const float* g1   = (const float*)d_in[6];
    const float* be1  = (const float*)d_in[7];
    const float* W2   = (const float*)d_in[8];
    const float* as2  = (const float*)d_in[9];
    const float* ad2  = (const float*)d_in[10];
    const float* b2   = (const float*)d_in[11];
    const float* g2   = (const float*)d_in[12];
    const float* be2  = (const float*)d_in[13];
    const float* Wa1  = (const float*)d_in[14];
    const float* ba1  = (const float*)d_in[15];
    const float* Wa2  = (const float*)d_in[16];
    const float* ba2  = (const float*)d_in[17];
    float* out = (float*)d_out;

    // workspace carve-up (float units)
    float* ws = (float*)d_ws;
    size_t o = 0;
    float*    scal   = ws + o;            o += 16;                     // [0]=enc(max), [1]=sumexp
    ushort_t* xbf    = (ushort_t*)(ws+o); o += (size_t)N_NODES*F_IN/2; // reused as x1bf later
    float*    hbuf   = ws + o;            o += (size_t)N_NODES*HD;     // h1 then h2
    float*    as_    = ws + o;            o += (size_t)N_NODES*2;
    float*    ad_    = ws + o;            o += (size_t)N_NODES*2;
    int*      amax   = (int*)(ws+o);      o += (size_t)N_NODES*2;
    float*    den    = ws + o;            o += (size_t)N_NODES*2;
    float*    agg    = ws + o;            o += (size_t)N_NODES*HD;
    float*    x1     = ws + o;            o += (size_t)N_NODES*HD;
    float*    x2     = ws + o;            o += (size_t)N_NODES*HD;
    float*    score  = ws + o;            o += (size_t)N_NODES;
    float*    ea     = ws + o;            o += (size_t)NET*2;
    ushort_t* w1frag = (ushort_t*)(ws+o); o += 8192/2;                  // 4*(128/32)*32*16 bf16
    ushort_t* w2frag = (ushort_t*)(ws+o); o += 4096/2;                  // 4*(64/32)*32*16 bf16

    const int TPB = 256;
    auto blks = [](long long n, int t) { return (int)((n + t - 1) / t); };

    // ---- converts / weight fragment packing ----
    cvt_bf_k<<<blks((long long)N_NODES*F_IN, TPB), TPB, 0, stream>>>(x, xbf, (long long)N_NODES*F_IN);
    pack_bfrag_k<<<2, TPB, 0, stream>>>(W1, w1frag, F_IN);  // 512 threads
    pack_bfrag_k<<<1, TPB, 0, stream>>>(W2, w2frag, HD);    // 256 threads

    const int gemm_blocks = blks((long long)(N_NODES / 16) * 32, TPB); // 6250 waves
    const int edge_blocks = blks(NET, TPB);
    const int msg_blocks  = blks((long long)NET * HD, TPB);
    const int node_wave_blocks = blks((long long)N_NODES * 32, TPB);

    // ================= Layer 1 =================
    gemm_rowpanel_wmma<F_IN><<<gemm_blocks, TPB, 0, stream>>>(xbf, w1frag, hbuf, N_NODES);
    att_dot_k<<<blks(N_NODES, TPB), TPB, 0, stream>>>(hbuf, as1, ad1, as_, ad_, N_NODES);

    fill_k<<<blks(N_NODES*2, TPB), TPB, 0, stream>>>((unsigned*)amax, 0x80000000u, N_NODES*2);
    fill_k<<<blks(N_NODES*2, TPB), TPB, 0, stream>>>((unsigned*)den, 0u, N_NODES*2);
    fill_k<<<blks((long long)N_NODES*HD, TPB), TPB, 0, stream>>>((unsigned*)agg, 0u, (long long)N_NODES*HD);

    edge_max_k<<<edge_blocks, TPB, 0, stream>>>(ei, as_, ad_, amax);
    edge_exp_k<<<edge_blocks, TPB, 0, stream>>>(ei, as_, ad_, amax, ea, den);
    edge_msg_k<<<msg_blocks, TPB, 0, stream>>>(ei, ea, den, hbuf, agg);

    ln_elu_k<<<node_wave_blocks, TPB, 0, stream>>>(agg, b1, g1, be1, nullptr, x1, xbf, N_NODES);

    // ================= Layer 2 =================
    gemm_rowpanel_wmma<HD><<<gemm_blocks, TPB, 0, stream>>>(xbf, w2frag, hbuf, N_NODES);
    att_dot_k<<<blks(N_NODES, TPB), TPB, 0, stream>>>(hbuf, as2, ad2, as_, ad_, N_NODES);

    fill_k<<<blks(N_NODES*2, TPB), TPB, 0, stream>>>((unsigned*)amax, 0x80000000u, N_NODES*2);
    fill_k<<<blks(N_NODES*2, TPB), TPB, 0, stream>>>((unsigned*)den, 0u, N_NODES*2);
    fill_k<<<blks((long long)N_NODES*HD, TPB), TPB, 0, stream>>>((unsigned*)agg, 0u, (long long)N_NODES*HD);

    edge_max_k<<<edge_blocks, TPB, 0, stream>>>(ei, as_, ad_, amax);
    edge_exp_k<<<edge_blocks, TPB, 0, stream>>>(ei, as_, ad_, amax, ea, den);
    edge_msg_k<<<msg_blocks, TPB, 0, stream>>>(ei, ea, den, hbuf, agg);

    ln_elu_k<<<node_wave_blocks, TPB, 0, stream>>>(agg, b2, g2, be2, x1, x2, nullptr, N_NODES);

    // ================= Pooling =================
    score_k<<<node_wave_blocks, TPB, 0, stream>>>(x2, Wa1, ba1, Wa2, ba2, score, N_NODES);

    fill_k<<<1, 16, 0, stream>>>((unsigned*)scal, 0x80000000u, 1);      // enc(-inf)
    fill_k<<<1, 16, 0, stream>>>((unsigned*)(scal + 1), 0u, 1);         // sumexp = 0
    fill_k<<<1, 64, 0, stream>>>((unsigned*)out, 0u, 64);

    smax_k<<<512, TPB, 0, stream>>>(score, (int*)scal, N_NODES);
    ssum_k<<<512, TPB, 0, stream>>>(score, (int*)scal, scal + 1, N_NODES);
    pool_k<<<512, 64, 0, stream>>>(x2, score, (int*)scal, scal + 1, out, N_NODES);
}
